// AttentionBlock_43361989820561
// MI455X (gfx1250) — matrix operand order repaired and checked by reading, not compiled, not run on hardware
//
#include <hip/hip_runtime.h>
#include <cstdint>

typedef __attribute__((ext_vector_type(16))) _Float16 v16h;
typedef __attribute__((ext_vector_type(8)))  float    v8f;

// Problem constants
static constexpr int BATCH  = 8;
static constexpr int CH     = 512;
static constexpr int L      = 1024;   // 32*32
static constexpr int HEADS  = 8;
static constexpr int CHH    = 64;     // CH / HEADS
static constexpr int GROUPS = 32;

// ---------------------------------------------------------------------------
// Kernel 1: GroupNorm (32 groups) -> f16 activations, stored TRANSPOSED as
// [b][l][c] so GEMM B-tiles stage with pure b128 copies (no LDS transpose).
// ---------------------------------------------------------------------------
__global__ void __launch_bounds__(256) gn_kernel(const float* __restrict__ x,
                                                 const float* __restrict__ nw,
                                                 const float* __restrict__ nb,
                                                 _Float16* __restrict__ xnT) {
    int b = blockIdx.x >> 5;         // / GROUPS
    int g = blockIdx.x & 31;
    int tid = threadIdx.x;
    const float4* base = (const float4*)(x + ((size_t)b * CH + g * 16) * L);
    float s = 0.f, sq = 0.f;
    for (int i = tid; i < 4096; i += 256) {     // 16 ch * 1024 = 4096 float4
        float4 v = base[i];
        s  += v.x + v.y + v.z + v.w;
        sq += v.x * v.x + v.y * v.y + v.z * v.z + v.w * v.w;
    }
    __shared__ float rs[256], rq[256];
    rs[tid] = s; rq[tid] = sq;
    __syncthreads();
    for (int off = 128; off > 0; off >>= 1) {
        if (tid < off) { rs[tid] += rs[tid + off]; rq[tid] += rq[tid + off]; }
        __syncthreads();
    }
    float mean = rs[0] * (1.f / 16384.f);
    float var  = rq[0] * (1.f / 16384.f) - mean * mean;
    float rinv = rsqrtf(var + 1e-5f);
    _Float16* dstT = xnT + (size_t)b * L * CH;
    for (int i = tid; i < 4096; i += 256) {
        float4 v = base[i];
        int c = g * 16 + (i >> 8);               // 256 float4 per channel
        int l = (i & 255) << 2;
        float w  = nw[c] * rinv;
        float bb = nb[c] - mean * w;
        dstT[(size_t)(l + 0) * CH + c] = (_Float16)(v.x * w + bb);
        dstT[(size_t)(l + 1) * CH + c] = (_Float16)(v.y * w + bb);
        dstT[(size_t)(l + 2) * CH + c] = (_Float16)(v.z * w + bb);
        dstT[(size_t)(l + 3) * CH + c] = (_Float16)(v.w * w + bb);
    }
}

// ---------------------------------------------------------------------------
// Kernel 2: convert weights f32 -> f16 (row-major, A operands)
// ---------------------------------------------------------------------------
__global__ void cvt_kernel(const float* __restrict__ qw, const float* __restrict__ pw,
                           _Float16* __restrict__ wq, _Float16* __restrict__ wp) {
    int i = blockIdx.x * 256 + threadIdx.x;
    if (i < 3 * CH * CH) wq[i] = (_Float16)qw[i];
    if (i < CH * CH)     wp[i] = (_Float16)pw[i];
}

// ---------------------------------------------------------------------------
// Shared WMMA GEMM mainloop, C(128x64) += A(MxK) * B(KxN).
// A row-major [m][k]; B given TRANSPOSED [n][k]. Ping-pong LDS buffers filled
// with CDNA5 async global->LDS b128 copies (ASYNCcnt); one barrier per K-step.
// 8 waves / 256-thread block, each wave owns a 16(M) x 64(N) strip.
// ---------------------------------------------------------------------------
__device__ __forceinline__ void gemm_tile(const _Float16* __restrict__ A,
                                          const _Float16* __restrict__ BnT,
                                          int K, int m0, int n0,
                                          _Float16 (*As)[128][32],
                                          _Float16 (*Bs)[64][32],
                                          v8f acc[4]) {
    int tid  = threadIdx.x;
    int lane = tid & 31, wv = tid >> 5;
    int ml = lane & 15, hh = lane >> 4;
    int row_a = tid >> 1, col_a = (tid & 1) << 4;   // 128 rows x 32 k
    int row_b = tid >> 2, col_b = (tid & 3) << 3;   // 64 rows x 32 k
    const _Float16* srcA = A   + (size_t)(m0 + row_a) * K + col_a;
    const _Float16* srcB = BnT + (size_t)(n0 + row_b) * K + col_b;

    auto issue = [&](int k0, int pn) {
        unsigned la = (unsigned)(uintptr_t)&As[pn][row_a][col_a];
        asm volatile("global_load_async_to_lds_b128 %0, %1, off"
                     :: "v"(la), "v"(srcA + k0) : "memory");
        asm volatile("global_load_async_to_lds_b128 %0, %1, off"
                     :: "v"(la + 16u), "v"(srcA + k0 + 8) : "memory");
        unsigned lb = (unsigned)(uintptr_t)&Bs[pn][row_b][col_b];
        asm volatile("global_load_async_to_lds_b128 %0, %1, off"
                     :: "v"(lb), "v"(srcB + k0) : "memory");
    };

    int p = 0;
    issue(0, 0);
    for (int k0 = 0; k0 < K; k0 += 32) {
        asm volatile("s_wait_asynccnt 0" ::: "memory");
        __syncthreads();
        if (k0 + 32 < K) issue(k0 + 32, p ^ 1);   // stream next tile during compute
        v16h af;
        {   // A fragment: lane(m=ml, half=hh): K chunks [8h,8h+8) and [16+8h,...)
            uint4* pa = (uint4*)&af;
            pa[0] = *(const uint4*)&As[p][wv * 16 + ml][8 * hh];
            pa[1] = *(const uint4*)&As[p][wv * 16 + ml][16 + 8 * hh];
        }
#pragma unroll
        for (int nf = 0; nf < 4; ++nf) {
            v16h bf;   // B fragment: lane(n=ml, half=hh): K chunks [8h,8h+8) and [16+8h,...)
            uint4* pb = (uint4*)&bf;
            pb[0] = *(const uint4*)&Bs[p][nf * 16 + ml][8 * hh];
            pb[1] = *(const uint4*)&Bs[p][nf * 16 + ml][16 + 8 * hh];
            acc[nf] = __builtin_amdgcn_wmma_f32_16x16x32_f16(
                false, af, false, bf, (short)0, acc[nf], false, false);
        }
        p ^= 1;
    }
}

// ---------------------------------------------------------------------------
// Kernel 3: qkv = qkv_w @ xn + qkv_b ; scatter into qT/kT (scaled, [b,h,l,c])
// and v ([b,h,c,l]) in f16
// ---------------------------------------------------------------------------
__global__ void __launch_bounds__(256) qkv_gemm_kernel(const _Float16* __restrict__ wq,
                                                       const _Float16* __restrict__ xnT,
                                                       const float* __restrict__ qkvb,
                                                       _Float16* __restrict__ qT,
                                                       _Float16* __restrict__ kT,
                                                       _Float16* __restrict__ vM) {
    __shared__ _Float16 As[2][128][32];
    __shared__ _Float16 Bs[2][64][32];
    int bid = blockIdx.x;
    int b   = bid / (12 * 16);
    int rem = bid % (12 * 16);
    int m0  = (rem / 16) * 128;
    int n0  = (rem % 16) * 64;
    v8f zero = {};
    v8f acc[4] = {zero, zero, zero, zero};
    gemm_tile(wq, xnT + (size_t)b * L * CH, CH, m0, n0, As, Bs, acc);

    int tid = threadIdx.x, lane = tid & 31, wv = tid >> 5;
    int ml = lane & 15, hh = lane >> 4;
    int o0    = m0 + wv * 16;
    int which = o0 >> 9;                  // 0=q, 1=k, 2=v
    int orem  = o0 & 511;
    int head  = orem >> 6;
    int cbase = (orem & 63) + 8 * hh;
    int obase = o0 + 8 * hh;
    const float SC = 0.35355339059327373f;   // 64^-0.25
#pragma unroll
    for (int nf = 0; nf < 4; ++nf) {
        int l = n0 + nf * 16 + ml;
        if (which < 2) {
            _Float16* dst = (which == 0 ? qT : kT)
                          + ((size_t)(b * HEADS + head) * L + l) * CHH + cbase;
            union { _Float16 h[8]; uint4 u; } pk;
#pragma unroll
            for (int r = 0; r < 8; ++r)
                pk.h[r] = (_Float16)((acc[nf][r] + qkvb[obase + r]) * SC);
            *(uint4*)dst = pk.u;
        } else {
#pragma unroll
            for (int r = 0; r < 8; ++r)
                vM[((size_t)(b * HEADS + head) * CHH + cbase + r) * L + l] =
                    (_Float16)(acc[nf][r] + qkvb[obase + r]);
        }
    }
}

// ---------------------------------------------------------------------------
// Kernel 4: flash attention, 64 heads of (64 x 1024). Block = 4 waves,
// each wave owns 16 query rows; online softmax over 32-column steps.
// V tiles stream into ping-pong LDS via async copies; P re-layout buffer is
// wave-private (DScnt ordering), so only one barrier per step.
// ---------------------------------------------------------------------------
__global__ void __launch_bounds__(128) attn_kernel(const _Float16* __restrict__ qT,
                                                   const _Float16* __restrict__ kT,
                                                   const _Float16* __restrict__ vM,
                                                   _Float16* __restrict__ aT) {
    __shared__ _Float16 vtile[2][64][32];   // V tile [c][s], shared by 4 waves
    __shared__ _Float16 pbuf[4][16][32];    // per-wave P staging (D-frag -> A-frag)
    int bh = blockIdx.x >> 4;               // batch*head
    int tb = blockIdx.x & 15;
    int tid = threadIdx.x, lane = tid & 31, wv = tid >> 5;
    int ml = lane & 15, hh = lane >> 4;
    int t0 = tb * 64 + wv * 16;
    int b = bh >> 3, head = bh & 7;

    // Q A-fragments (K = 64 split into two K=32 fragments), scale pre-folded
    const _Float16* qrow = qT + ((size_t)bh * L + t0 + ml) * CHH;
    v16h qa[2];
#pragma unroll
    for (int f = 0; f < 2; ++f) {
        uint4* pa = (uint4*)&qa[f];
        pa[0] = *(const uint4*)(qrow + f * 32 + 8 * hh);
        pa[1] = *(const uint4*)(qrow + f * 32 + 16 + 8 * hh);
    }

    // async V staging: each thread copies 32B of the 64x32 tile
    int c_v = tid >> 1, col_v = (tid & 1) << 4;
    const _Float16* vsrc = vM + ((size_t)bh * CHH + c_v) * L + col_v;
    auto issue_v = [&](int s0n, int pn) {
        unsigned la = (unsigned)(uintptr_t)&vtile[pn][c_v][col_v];
        asm volatile("global_load_async_to_lds_b128 %0, %1, off"
                     :: "v"(la), "v"(vsrc + s0n) : "memory");
        asm volatile("global_load_async_to_lds_b128 %0, %1, off"
                     :: "v"(la + 16u), "v"(vsrc + s0n + 8) : "memory");
    };

    float mx[8], ls[8];
    v8f zero = {};
    v8f accO[4] = {zero, zero, zero, zero};
#pragma unroll
    for (int r = 0; r < 8; ++r) { mx[r] = -1e30f; ls[r] = 0.f; }

    int p = 0;
    issue_v(0, 0);
    for (int s0 = 0; s0 < L; s0 += 32) {
        asm volatile("s_wait_asynccnt 0" ::: "memory");
        __syncthreads();
        if (s0 + 32 < L) issue_v(s0 + 32, p ^ 1);

        // S strip (16 x 32): two 16x16 tiles, K = 64 via chained WMMA
        v8f sa[2];
#pragma unroll
        for (int ct = 0; ct < 2; ++ct) {
            const _Float16* krow = kT + ((size_t)bh * L + s0 + ct * 16 + ml) * CHH;
            v16h kb;
            uint4* pb = (uint4*)&kb;
            pb[0] = *(const uint4*)(krow + 8 * hh);
            pb[1] = *(const uint4*)(krow + 16 + 8 * hh);
            v8f z = {};
            z = __builtin_amdgcn_wmma_f32_16x16x32_f16(false, qa[0], false, kb,
                                                       (short)0, z, false, false);
            pb[0] = *(const uint4*)(krow + 32 + 8 * hh);
            pb[1] = *(const uint4*)(krow + 32 + 16 + 8 * hh);
            sa[ct] = __builtin_amdgcn_wmma_f32_16x16x32_f16(false, qa[1], false, kb,
                                                            (short)0, z, false, false);
        }
        // online softmax (f32), write P (f16) to wave-private LDS
#pragma unroll
        for (int r = 0; r < 8; ++r) {
            float v0 = sa[0][r], v1 = sa[1][r];
            float mloc = fmaxf(v0, v1);
            mloc = fmaxf(mloc, __shfl_xor(mloc, 1));
            mloc = fmaxf(mloc, __shfl_xor(mloc, 2));
            mloc = fmaxf(mloc, __shfl_xor(mloc, 4));
            mloc = fmaxf(mloc, __shfl_xor(mloc, 8));
            float mn   = fmaxf(mx[r], mloc);
            float corr = __expf(mx[r] - mn);
            mx[r] = mn;
            float p0 = __expf(v0 - mn), p1 = __expf(v1 - mn);
            float rsum = p0 + p1;
            rsum += __shfl_xor(rsum, 1);
            rsum += __shfl_xor(rsum, 2);
            rsum += __shfl_xor(rsum, 4);
            rsum += __shfl_xor(rsum, 8);
            ls[r] = ls[r] * corr + rsum;
#pragma unroll
            for (int cf = 0; cf < 4; ++cf) accO[cf][r] *= corr;
            pbuf[wv][8 * hh + r][ml]      = (_Float16)p0;
            pbuf[wv][8 * hh + r][ml + 16] = (_Float16)p1;
        }
        // O += P(16x32) x V'(32x64): read P back as an A-fragment (wave-local)
        v16h pf;
        {
            uint4* pp = (uint4*)&pf;
            pp[0] = *(const uint4*)&pbuf[wv][ml][8 * hh];
            pp[1] = *(const uint4*)&pbuf[wv][ml][16 + 8 * hh];
        }
#pragma unroll
        for (int cf = 0; cf < 4; ++cf) {
            v16h vb;
            uint4* pv = (uint4*)&vb;
            pv[0] = *(const uint4*)&vtile[p][cf * 16 + ml][8 * hh];
            pv[1] = *(const uint4*)&vtile[p][cf * 16 + ml][16 + 8 * hh];
            accO[cf] = __builtin_amdgcn_wmma_f32_16x16x32_f16(false, pf, false, vb,
                                                              (short)0, accO[cf],
                                                              false, false);
        }
        p ^= 1;
    }
    // normalize; store attention output TRANSPOSED [b][l][c] for proj GEMM
#pragma unroll
    for (int cf = 0; cf < 4; ++cf) {
        int cg = head * CHH + cf * 16 + ml;
#pragma unroll
        for (int r = 0; r < 8; ++r)
            aT[((size_t)b * L + t0 + 8 * hh + r) * CH + cg] =
                (_Float16)(accO[cf][r] / ls[r]);
    }
}

// ---------------------------------------------------------------------------
// Kernel 5: out = proj_w @ a + proj_b + x (residual), fp32 output
// ---------------------------------------------------------------------------
__global__ void __launch_bounds__(256) proj_gemm_kernel(const _Float16* __restrict__ wp,
                                                        const _Float16* __restrict__ aT,
                                                        const float* __restrict__ projb,
                                                        const float* __restrict__ x,
                                                        float* __restrict__ out) {
    __shared__ _Float16 As[2][128][32];
    __shared__ _Float16 Bs[2][64][32];
    int bid = blockIdx.x;
    int b   = bid / (4 * 16);
    int rem = bid % (4 * 16);
    int m0  = (rem / 16) * 128;
    int n0  = (rem % 16) * 64;
    v8f zero = {};
    v8f acc[4] = {zero, zero, zero, zero};
    gemm_tile(wp, aT + (size_t)b * L * CH, CH, m0, n0, As, Bs, acc);

    int tid = threadIdx.x, lane = tid & 31, wv = tid >> 5;
    int ml = lane & 15, hh = lane >> 4;
    int o0 = m0 + wv * 16 + 8 * hh;
#pragma unroll
    for (int nf = 0; nf < 4; ++nf) {
        int l = n0 + nf * 16 + ml;
#pragma unroll
        for (int r = 0; r < 8; ++r) {
            int o = o0 + r;
            size_t idx = ((size_t)b * CH + o) * L + l;
            out[idx] = acc[nf][r] + projb[o] + x[idx];
        }
    }
}

// ---------------------------------------------------------------------------
// Host launcher
// ---------------------------------------------------------------------------
extern "C" void kernel_launch(void* const* d_in, const int* in_sizes, int n_in,
                              void* d_out, int out_size, void* d_ws, size_t ws_size,
                              hipStream_t stream) {
    const float* x      = (const float*)d_in[0];
    const float* norm_w = (const float*)d_in[1];
    const float* norm_b = (const float*)d_in[2];
    const float* qkv_w  = (const float*)d_in[3];
    const float* qkv_b  = (const float*)d_in[4];
    const float* proj_w = (const float*)d_in[5];
    const float* proj_b = (const float*)d_in[6];
    float* out = (float*)d_out;

    // workspace layout (bytes)
    char* ws = (char*)d_ws;
    size_t off = 0;
    _Float16* xnT = (_Float16*)(ws + off); off += (size_t)BATCH * CH * L * 2;  // [b][l][c]
    _Float16* wq  = (_Float16*)(ws + off); off += (size_t)3 * CH * CH * 2;
    _Float16* wp  = (_Float16*)(ws + off); off += (size_t)CH * CH * 2;
    _Float16* qT  = (_Float16*)(ws + off); off += (size_t)BATCH * CH * L * 2;  // [b,h,l,c]
    _Float16* kT  = (_Float16*)(ws + off); off += (size_t)BATCH * CH * L * 2;  // [b,h,l,c]
    _Float16* vM  = (_Float16*)(ws + off); off += (size_t)BATCH * CH * L * 2;  // [b,h,c,l]
    _Float16* aT  = (_Float16*)(ws + off); off += (size_t)BATCH * CH * L * 2;  // [b][l][c]

    gn_kernel<<<BATCH * GROUPS, 256, 0, stream>>>(x, norm_w, norm_b, xnT);
    cvt_kernel<<<(3 * CH * CH + 255) / 256, 256, 0, stream>>>(qkv_w, proj_w, wq, wp);
    qkv_gemm_kernel<<<BATCH * 12 * 16, 256, 0, stream>>>(wq, xnT, qkv_b, qT, kT, vM);
    attn_kernel<<<BATCH * HEADS * (L / 64), 128, 0, stream>>>(qT, kT, vM, aT);
    proj_gemm_kernel<<<BATCH * 4 * 16, 256, 0, stream>>>(wp, aT, proj_b, x, out);
}
